// Binary_conv_bn_relu_38414187495897
// MI455X (gfx1250) — 8-bit matrix operand order repaired and checked by reading, compiled, not run on hardware
//
#include <hip/hip_runtime.h>

typedef int   v16i __attribute__((ext_vector_type(16)));
typedef int   v4i  __attribute__((ext_vector_type(4)));
typedef float v8f  __attribute__((ext_vector_type(8)));

#define CIN   256
#define COUT  256
#define Hs    56
#define Ws    56
#define NB    64
#define HW    (Hs*Ws)
#define NPIX  (NB*HW)
#define KCH   18
#define OH_BLK 4
#define AROWS (18*(OH_BLK+2))
#define AROW_STRIDE_B 272

#define FP8_P1 ((unsigned char)0x38)
#define FP8_M1 ((unsigned char)0xB8)
#define FP8_Z  ((unsigned char)0x00)

__global__ void bn_stats_kernel(const float* __restrict__ x,
                                const float* __restrict__ gamma,
                                const float* __restrict__ beta,
                                float* __restrict__ scale,
                                float* __restrict__ shift) {
    __shared__ float ssum[256];
    __shared__ float ssq[256];
    int c = blockIdx.x;
    int tid = threadIdx.x;
    float sum = 0.f, sq = 0.f;
    const float* xc = x + (size_t)c * HW;
    for (int i = tid; i < NPIX; i += 256) {
        int n = i / HW, p = i - n * HW;
        float v = xc[(size_t)n * CIN * HW + p];
        sum += v; sq += v * v;
    }
    ssum[tid] = sum; ssq[tid] = sq;
    __syncthreads();
    for (int s = 128; s > 0; s >>= 1) {
        if (tid < s) { ssum[tid] += ssum[tid + s]; ssq[tid] += ssq[tid + s]; }
        __syncthreads();
    }
    if (tid == 0) {
        float mean = ssum[0] / (float)NPIX;
        float var  = ssq[0] / (float)NPIX - mean * mean;
        float s = gamma[c] * rsqrtf(var + 1e-5f);
        scale[c] = s;
        shift[c] = beta[c] - mean * s;
    }
}

__global__ void wrepack_kernel(const float* __restrict__ w,
                               unsigned char* __restrict__ wfrag) {
    int idx = blockIdx.x * blockDim.x + threadIdx.x;
    if (idx >= 16 * KCH * 32) return;
    int lane = idx & 31;
    int kc   = (idx >> 5) % KCH;
    int cot  = idx / (32 * KCH);
    int khw = kc >> 1;
    int kh = khw / 3, kw = khw % 3;
    int ci0 = (kc & 1) * 128;
    int co  = cot * 16 + (lane & 15);
    int hi16 = (lane >= 16) ? 16 : 0;
    unsigned char* dst = wfrag + (size_t)idx * 64;
    for (int j = 0; j < 16; ++j) {
        int kbase = (j >> 2) * 32 + hi16 + (j & 3) * 4;
        for (int b = 0; b < 4; ++b) {
            int ci = ci0 + kbase + b;
            float wv = w[(((size_t)co * CIN + ci) * 3 + kh) * 3 + kw];
            dst[j * 4 + b] = (wv > 0.f) ? FP8_P1 : ((wv < 0.f) ? FP8_M1 : FP8_Z);
        }
    }
}

__global__ void binarize_kernel(const float* __restrict__ x,
                                const float* __restrict__ scale,
                                const float* __restrict__ shift,
                                unsigned char* __restrict__ xb) {
    __shared__ unsigned char tile[16][17];
    int hw0 = blockIdx.x * 16;
    int c0  = blockIdx.y * 16;
    int n   = blockIdx.z;
    int tx = threadIdx.x, ty = threadIdx.y;
    int c = c0 + ty, p = hw0 + tx;
    float v = x[((size_t)n * CIN + c) * HW + p] * scale[c] + shift[c];
    tile[ty][tx] = (v > 0.f) ? FP8_P1 : ((v < 0.f) ? FP8_M1 : FP8_Z);
    __syncthreads();
    xb[((size_t)n * HW + hw0 + ty) * CIN + c0 + tx] = tile[tx][ty];
}

__global__ void __launch_bounds__(256)
bconv_wmma_kernel(const unsigned char* __restrict__ xb,
                  const unsigned char* __restrict__ wfrag,
                  const float* __restrict__ bias,
                  float* __restrict__ out) {
    __shared__ int4  aldsv[AROWS * 17];
    __shared__ float otile[8][16 * 20];

    int bid    = blockIdx.x;
    int cohalf = bid & 1;
    int pb     = bid >> 1;
    int ow0 = (pb & 3) * 16;
    int t2  = pb >> 2;
    int oh0 = (t2 % 14) * OH_BLK;
    int n   = t2 / 14;

    int tid  = threadIdx.x;
    int wid  = tid >> 5;
    int lane = tid & 31;
    int cot  = cohalf * 8 + wid;

    for (int t = tid; t < AROWS * 16; t += 256) {
        int r  = t >> 4;
        int cb = t & 15;
        int krow = r / 18;
        int col  = r - krow * 18;
        int ih = oh0 + krow - 1;
        int iw = ow0 + col - 1;
        int4 val = make_int4(0, 0, 0, 0);
        if (ih >= 0 && ih < Hs && iw >= 0 && iw < Ws) {
            val = *(const int4*)(xb + (((size_t)n * Hs + ih) * Ws + iw) * CIN + cb * 16);
        }
        aldsv[r * 17 + cb] = val;
    }
    __syncthreads();

    int m   = lane & 15;
    int hiK = lane >> 4;

    v8f acc[OH_BLK];
    #pragma unroll
    for (int t = 0; t < OH_BLK; ++t) acc[t] = (v8f){0.f,0.f,0.f,0.f,0.f,0.f,0.f,0.f};

    const unsigned char* alds = (const unsigned char*)aldsv;
    const v4i* wf = (const v4i*)(wfrag + ((size_t)cot * KCH * 32 + lane) * 64);

    for (int khw = 0; khw < 9; ++khw) {
        int kh = khw / 3, kw = khw - 3 * (khw / 3);
        #pragma unroll
        for (int c2 = 0; c2 < 2; ++c2) {
            int kc = khw * 2 + c2;
            v16i b;
            #pragma unroll
            for (int q = 0; q < 4; ++q) {
                v4i bq = wf[(size_t)kc * 128 + q];
                b[4*q+0] = bq[0]; b[4*q+1] = bq[1]; b[4*q+2] = bq[2]; b[4*q+3] = bq[3];
            }
            #pragma unroll
            for (int t = 0; t < OH_BLK; ++t) {
                const unsigned char* arow =
                    alds + ((t + kh) * 18 + m + kw) * AROW_STRIDE_B + c2 * 128 + hiK * 16;
                v16i a;
                #pragma unroll
                for (int h = 0; h < 2; ++h) {
                    #pragma unroll
                    for (int p = 0; p < 4; ++p) {
                        int2 d = *(const int2*)(arow + h * 64 + (p >> 1) * 32 + (p & 1) * 8);
                        a[h * 8 + 2 * p]     = d.x;
                        a[h * 8 + 2 * p + 1] = d.y;
                    }
                }
                acc[t] = __builtin_amdgcn_wmma_f32_16x16x128_fp8_fp8(
                            a, b, (short)0, acc[t], false, false);
            }
        }
    }

    int co_n = lane & 15;
    float bv = bias[cot * 16 + co_n];
    int co_r = lane >> 1;
    int half = lane & 1;
    bool storeok = (ow0 + half * 8) < Ws;

    #pragma unroll
    for (int t = 0; t < OH_BLK; ++t) {
        #pragma unroll
        for (int v = 0; v < 8; ++v) {
            float r = acc[t][v] + bv;
            r = r > 0.f ? r : 0.f;
            otile[wid][co_n * 20 + v + 8 * hiK] = r;
        }
        if (storeok) {
            const float* row = &otile[wid][co_r * 20 + half * 8];
            float4 r0 = *(const float4*)(row);
            float4 r1 = *(const float4*)(row + 4);
            float* o = out + (((size_t)n * COUT + cot * 16 + co_r) * Hs + oh0 + t) * Ws
                           + ow0 + half * 8;
            *(float4*)(o)     = r0;
            *(float4*)(o + 4) = r1;
        }
    }
}

extern "C" void kernel_launch(void* const* d_in, const int* in_sizes, int n_in,
                              void* d_out, int out_size, void* d_ws, size_t ws_size,
                              hipStream_t stream) {
    const float* x     = (const float*)d_in[0];
    const float* gamma = (const float*)d_in[1];
    const float* beta  = (const float*)d_in[2];
    const float* w     = (const float*)d_in[3];
    const float* b     = (const float*)d_in[4];
    float* out = (float*)d_out;

    char* ws = (char*)d_ws;
    float*         scale = (float*)(ws);
    float*         shift = (float*)(ws + 1024);
    unsigned char* wfrag = (unsigned char*)(ws + 4096);
    unsigned char* xb    = (unsigned char*)(ws + (1u << 20));

    bn_stats_kernel<<<CIN, 256, 0, stream>>>(x, gamma, beta, scale, shift);
    wrepack_kernel<<<(16 * KCH * 32 + 255) / 256, 256, 0, stream>>>(w, wfrag);
    dim3 bgrid(HW / 16, CIN / 16, NB);
    binarize_kernel<<<bgrid, dim3(16, 16), 0, stream>>>(x, scale, shift, xb);

    int nblocks = 4 * (Hs / OH_BLK) * NB * 2;
    bconv_wmma_kernel<<<nblocks, 256, 0, stream>>>(xb, wfrag, b, out);
}
